// VectorQuantizerEMA_8022998909237
// MI455X (gfx1250) — matrix operand order repaired and checked by reading, not compiled, not run on hardware
//
#include <hip/hip_runtime.h>
#include <hip/hip_bf16.h>
#include <stdint.h>

// ---------------- problem constants ----------------
constexpr int   D     = 256;       // embedding dim
constexpr int   E     = 1024;      // number of embeddings
constexpr int   N     = 64 * 512;  // flattened rows = 32768
constexpr float DECAY = 0.99f;
constexpr float EPS_  = 1e-5f;
constexpr float CCOST = 0.5f;

// ---------------- d_out layout (floats, reference return order) ----------------
constexpr long long ZQ_OFF   = 0;
constexpr long long LOSS_OFF = (long long)N * D;           // 8388608
constexpr long long IDX_OFF  = LOSS_OFF + 1;
constexpr long long NCB_OFF  = IDX_OFF + N;
constexpr long long NCS_OFF  = NCB_OFF + (long long)E * D;
constexpr long long NES_OFF  = NCS_OFF + E;

// ---------------- workspace layout (bytes); total ~2.24 MB ----------------
constexpr size_t WS_CB   = 0;                   // bf16 codebook, WMMA-B tile layout: 64 tiles * 16KB (hi 8KB | lo 8KB)
constexpr size_t WS_CSQ  = 1048576;             // float[E]
constexpr size_t WS_CNT  = WS_CSQ + 4096;       // float[E]     batch cluster size accum
constexpr size_t WS_EMB  = WS_CNT + 4096;       // float[E*D]   batch embed sum accum
constexpr size_t WS_LOSS = WS_EMB + 1048576;    // float[1]
constexpr size_t WS_NVAL = WS_LOSS + 4;         // float[1]
constexpr size_t WS_IDX  = WS_NVAL + 4;         // int[N]

// ---------------- vector types ----------------
typedef float    v4f   __attribute__((ext_vector_type(4)));
typedef float    v8f   __attribute__((ext_vector_type(8)));
typedef unsigned v4u   __attribute__((ext_vector_type(4)));
typedef unsigned v8u   __attribute__((ext_vector_type(8)));
typedef short    v16s  __attribute__((ext_vector_type(16)));
typedef __bf16   v16bf __attribute__((ext_vector_type(16)));
// exact parameter type of the async-to-LDS builtin (gcc-style int vector)
typedef int      g4i   __attribute__((vector_size(16)));

// ---------------- bf16 helpers (pure bit math; RNE) ----------------
static __device__ __forceinline__ unsigned short f2bf(float f) {
  unsigned u = __builtin_bit_cast(unsigned, f);
  unsigned r = u + 0x7fffu + ((u >> 16) & 1u);
  return (unsigned short)(r >> 16);
}
static __device__ __forceinline__ float bf2f(unsigned short s) {
  return __builtin_bit_cast(float, (unsigned)s << 16);
}

static __device__ __forceinline__ v8f wmma_bf16(v16bf a, v16bf b, v8f c) {
  // (neg_a, A, neg_b, B, c_mod, C, reuse_a, reuse_b)
  return __builtin_amdgcn_wmma_f32_16x16x32_bf16(false, a, false, b, (short)0, c,
                                                 false, false);
}

// ---------------- async global->LDS copy (CDNA5 path, guarded) ----------------
#if (__has_builtin(__builtin_amdgcn_global_load_async_to_lds_b128) && \
     __has_builtin(__builtin_amdgcn_s_wait_asynccnt))
#define VQ_ASYNC_LDS 1
#else
#define VQ_ASYNC_LDS 0
#endif

static __device__ __forceinline__ void cp16_g2l(v4u* lds, const v4u* g) {
#if VQ_ASYNC_LDS
  __builtin_amdgcn_global_load_async_to_lds_b128(
      (__attribute__((address_space(1))) g4i*)g,
      (__attribute__((address_space(3))) g4i*)lds, 0, 0);
#else
  *lds = *g;  // fallback: global_load_b128 + ds_store_b128
#endif
}
static __device__ __forceinline__ void wait_g2l() {
#if VQ_ASYNC_LDS
  __builtin_amdgcn_s_wait_asynccnt(0);
#endif
}

// ==================================================================
// Kernel 0: zero accumulators (counts, embed sums, loss, n)
// ==================================================================
__global__ void vq_init_zero_kernel(float* __restrict__ p, int n) {
  int i = blockIdx.x * 256 + threadIdx.x;
  if (i < n) p[i] = 0.0f;
}

// ==================================================================
// Kernel 1: codebook prep.  bf16 hi/lo split into WMMA-B lane layout
// + c_sq row reduction.  One block (256 thr) per codebook row.
// B 32x16 bf16 layout: lane = h*16+n  (n = e%16, h from K%32),
// element j of lane = K = kc*32 + 16h + j, contiguous per lane.
// ==================================================================
__global__ __launch_bounds__(256) void vq_prep_codebook_kernel(
    const float* __restrict__ cb, unsigned short* __restrict__ cbB,
    float* __restrict__ csq) {
  __shared__ float red[256];
  const int e = blockIdx.x;
  const int k = threadIdx.x;
  const float x = cb[(size_t)e * D + k];
  const unsigned short hi = f2bf(x);
  const unsigned short lo = f2bf(x - bf2f(hi));
  const int etile = e >> 4, nn = e & 15;
  const int kc = k >> 5, r = k & 31, h = r >> 4, j = r & 15;
  const int lane = (h << 4) | nn;
  unsigned short* tile = cbB + (size_t)etile * 8192;  // 8192 u16 = 16KB per tile
  const int o = (kc * 32 + lane) * 16 + j;
  tile[o] = hi;
  tile[4096 + o] = lo;
  red[k] = x * x;
  __syncthreads();
  for (int s = 128; s > 0; s >>= 1) {
    if (k < s) red[k] += red[k + s];
    __syncthreads();
  }
  if (k == 0) csq[e] = red[0];
}

// ==================================================================
// Kernel 2: distances via bf16 WMMA (hi/lo split, 3 terms) + argmin.
// 256 threads = 8 waves; wave owns a 16-row A tile in registers.
// B tiles double-buffered in LDS via async global->LDS loads.
// ==================================================================
__global__ __launch_bounds__(256) void vq_argmin_kernel(
    const float* __restrict__ z, const unsigned char* __restrict__ cbB,
    const float* __restrict__ csq, int* __restrict__ idx_out,
    float* __restrict__ out) {
  __shared__ v8u Bbuf[2][512];  // 2 x 16KB (hi chunks 0..255, lo chunks 256..511)
  const int tid  = threadIdx.x;
  const int wv   = tid >> 5;
  const int lane = tid & 31;
  const int nn   = lane & 15;
  const int h    = lane >> 4;
  const int rowBase = blockIdx.x * 128 + wv * 16;
  const int m       = rowBase + nn;

  // ---- build A operands (16 rows x 256 K) in bf16 hi/lo WMMA layout ----
  // A 16x32 layout: lane (m=l%16, h=l/16) elems 0..15 = K in [16h,16h+16)
  // (contiguous) within each 32-wide K chunk.
  const v4f* zr = (const v4f*)(z + (size_t)m * D);
  v16bf Ahi[8], Alo[8];
  float zp = 0.0f;
#pragma unroll
  for (int kc = 0; kc < 8; ++kc) {
    v16s hs, ls;
#pragma unroll
    for (int p = 0; p < 2; ++p) {
      const int base = kc * 32 + 16 * h + p * 8;
      v4f f0 = zr[base >> 2];
      v4f f1 = zr[(base >> 2) + 1];
#pragma unroll
      for (int q = 0; q < 4; ++q) {
        float x0 = f0[q], x1 = f1[q];
        unsigned short h0 = f2bf(x0), h1 = f2bf(x1);
        hs[p * 8 + q]     = (short)h0;
        hs[p * 8 + 4 + q] = (short)h1;
        ls[p * 8 + q]     = (short)f2bf(x0 - bf2f(h0));
        ls[p * 8 + 4 + q] = (short)f2bf(x1 - bf2f(h1));
        zp += x0 * x0 + x1 * x1;
      }
    }
    Ahi[kc] = __builtin_bit_cast(v16bf, hs);
    Alo[kc] = __builtin_bit_cast(v16bf, ls);
  }
  // row sum-of-squares: lanes l and l^16 jointly cover a full row
  float zsq = zp + __shfl_xor(zp, 16, 32);
  float zs8[8];
#pragma unroll
  for (int j = 0; j < 8; ++j) zs8[j] = __shfl(zsq, j + 8 * h, 16);

  float best[8];
  int   besti[8];
#pragma unroll
  for (int j = 0; j < 8; ++j) { best[j] = 3.0e38f; besti[j] = 0; }

  // ---- issue first B tile copy (16KB: 4 x b128 per thread) ----
  {
    const v4u* src = (const v4u*)(cbB);
    v4u* dst = (v4u*)&Bbuf[0][0];
#pragma unroll
    for (int q = 0; q < 4; ++q) cp16_g2l(&dst[tid * 4 + q], &src[tid * 4 + q]);
  }

  for (int et = 0; et < E / 16; ++et) {
    wait_g2l();
    __syncthreads();  // tile 'et' resident in Bbuf[et&1] for all waves
    if (et + 1 < E / 16) {
      const v4u* src = (const v4u*)(cbB + (size_t)(et + 1) * 16384);
      v4u* dst = (v4u*)&Bbuf[(et + 1) & 1][0];
#pragma unroll
      for (int q = 0; q < 4; ++q) cp16_g2l(&dst[tid * 4 + q], &src[tid * 4 + q]);
    }
    const int b = et & 1;
    v8f acc = {0.f, 0.f, 0.f, 0.f, 0.f, 0.f, 0.f, 0.f};
#pragma unroll
    for (int kc = 0; kc < 8; ++kc) {
      const int ci = kc * 32 + lane;
      v16bf Bh = __builtin_bit_cast(v16bf, Bbuf[b][ci]);
      v16bf Bl = __builtin_bit_cast(v16bf, Bbuf[b][256 + ci]);
      acc = wmma_bf16(Ahi[kc], Bh, acc);   // hi*hi
      acc = wmma_bf16(Alo[kc], Bh, acc);   // lo*hi
      acc = wmma_bf16(Ahi[kc], Bl, acc);   // hi*lo
    }
    // C layout: acc[j] is element (M = j + 8h, code n = lane%16)
    const float cs = csq[et * 16 + nn];
    const int   ec = et * 16 + nn;
#pragma unroll
    for (int j = 0; j < 8; ++j) {
      float dist = zs8[j] + cs - 2.0f * acc[j];
      if (dist < best[j]) { best[j] = dist; besti[j] = ec; }
    }
  }

  // cross-lane argmin over the 16 lanes sharing each row (tie -> smaller index)
#pragma unroll
  for (int j = 0; j < 8; ++j) {
#pragma unroll
    for (int off = 8; off >= 1; off >>= 1) {
      float od = __shfl_xor(best[j], off, 16);
      int   oi = __shfl_xor(besti[j], off, 16);
      if (od < best[j] || (od == best[j] && oi < besti[j])) {
        best[j] = od; besti[j] = oi;
      }
    }
  }
  if (nn == 0) {
#pragma unroll
    for (int j = 0; j < 8; ++j) {
      const int row = rowBase + j + 8 * h;
      idx_out[row] = besti[j];
      out[IDX_OFF + row] = (float)besti[j];
    }
  }
}

// ==================================================================
// Kernel 3: gather z_q (old codebook), commitment loss, EMA scatter
// ==================================================================
__global__ __launch_bounds__(256) void vq_gather_stats_kernel(
    const float* __restrict__ z, const float* __restrict__ cb,
    const int* __restrict__ idx, float* __restrict__ out,
    float* __restrict__ emb, float* __restrict__ cnt,
    float* __restrict__ loss) {
  __shared__ float red[256];
  const int t = threadIdx.x;  // = d (0..255)
  float ls = 0.0f;
  for (int i = 0; i < 32; ++i) {
    const int r = blockIdx.x * 32 + i;
    const int e = idx[r];
    const float q = cb[(size_t)e * D + t];
    const float x = z[(size_t)r * D + t];
    out[ZQ_OFF + (size_t)r * D + t] = q;  // straight-through value == gather
    const float df = x - q;
    ls += df * df;
    atomicAdd(&emb[(size_t)e * D + t], x);
    if (t == 0) atomicAdd(&cnt[e], 1.0f);
  }
  red[t] = ls;
  __syncthreads();
  for (int s = 128; s > 0; s >>= 1) {
    if (t < s) red[t] += red[t + s];
    __syncthreads();
  }
  if (t == 0) atomicAdd(loss, red[0]);
}

// ==================================================================
// Kernel 4: new_cluster_size, n = sum, vq_loss
// ==================================================================
__global__ __launch_bounds__(256) void vq_finalize_cs_kernel(
    const float* __restrict__ ecs, const float* __restrict__ cnt,
    const float* __restrict__ loss, float* __restrict__ out,
    float* __restrict__ nval) {
  __shared__ float red[256];
  const int t = threadIdx.x;
  float s = 0.0f;
  for (int e = t; e < E; e += 256) {
    float v = DECAY * ecs[e] + (1.0f - DECAY) * cnt[e];
    out[NCS_OFF + e] = v;
    s += v;
  }
  red[t] = s;
  __syncthreads();
  for (int st = 128; st > 0; st >>= 1) {
    if (t < st) red[t] += red[t + st];
    __syncthreads();
  }
  if (t == 0) {
    nval[0] = red[0];
    out[LOSS_OFF] = CCOST * loss[0] / (float)((long long)N * D);
  }
}

// ==================================================================
// Kernel 5: new_embed_sum + smoothed new_codebook
// ==================================================================
__global__ __launch_bounds__(256) void vq_finalize_embed_kernel(
    const float* __restrict__ ees, const float* __restrict__ emb,
    const float* __restrict__ nval, float* __restrict__ out) {
  const int i = blockIdx.x * 256 + threadIdx.x;  // < E*D
  const int e = i >> 8;
  const float nes = DECAY * ees[i] + (1.0f - DECAY) * emb[i];
  out[NES_OFF + i] = nes;
  const float ncs = out[NCS_OFF + e];
  const float n = nval[0];
  const float sm = (ncs + EPS_) / (n + (float)E * EPS_) * n;
  out[NCB_OFF + i] = nes / sm;
}

// ==================================================================
extern "C" void kernel_launch(void* const* d_in, const int* in_sizes, int n_in,
                              void* d_out, int out_size, void* d_ws,
                              size_t ws_size, hipStream_t stream) {
  const float* z   = (const float*)d_in[0];
  const float* cb  = (const float*)d_in[1];
  const float* ecs = (const float*)d_in[2];
  const float* ees = (const float*)d_in[3];
  float* out = (float*)d_out;
  unsigned char* ws = (unsigned char*)d_ws;

  unsigned short* cbB  = (unsigned short*)(ws + WS_CB);
  float* csq  = (float*)(ws + WS_CSQ);
  float* cnt  = (float*)(ws + WS_CNT);
  float* emb  = (float*)(ws + WS_EMB);
  float* loss = (float*)(ws + WS_LOSS);
  float* nval = (float*)(ws + WS_NVAL);
  int*   idx  = (int*)(ws + WS_IDX);

  // zero counts + embed accum + loss + n (contiguous region)
  const int nz = (4096 + 1048576 + 8) / 4;  // 263170 floats
  vq_init_zero_kernel<<<(nz + 255) / 256, 256, 0, stream>>>(cnt, nz);
  vq_prep_codebook_kernel<<<E, 256, 0, stream>>>(cb, cbB, csq);
  vq_argmin_kernel<<<N / 128, 256, 0, stream>>>(z, (const unsigned char*)cbB,
                                                csq, idx, out);
  vq_gather_stats_kernel<<<N / 32, 256, 0, stream>>>(z, cb, idx, out, emb, cnt,
                                                     loss);
  vq_finalize_cs_kernel<<<1, 256, 0, stream>>>(ecs, cnt, loss, out, nval);
  vq_finalize_embed_kernel<<<(E * D) / 256, 256, 0, stream>>>(ees, emb, nval,
                                                              out);
}
